// LancetBlockFull_56049323213100
// MI455X (gfx1250) — matrix operand order repaired and checked by reading, not compiled, not run on hardware
//
#include <hip/hip_runtime.h>
#include <hip/hip_bf16.h>

typedef __attribute__((ext_vector_type(16))) __bf16 v16bf;
typedef __attribute__((ext_vector_type(8)))  __bf16 v8bf;
typedef __attribute__((ext_vector_type(8)))  float  v8f;

#define WMMA_BF16(a, b, c) \
  __builtin_amdgcn_wmma_f32_16x16x32_bf16(false, (a), false, (b), (short)0, (c), false, false)

__device__ __forceinline__ v16bf cat8(v8bf lo, v8bf hi) {
  return __builtin_shufflevector(lo, hi, 0,1,2,3,4,5,6,7,8,9,10,11,12,13,14,15);
}

__device__ __forceinline__ v8bf cvt8(float4 f0, float4 f1) {
  v8bf h;
  h[0] = (__bf16)f0.x; h[1] = (__bf16)f0.y; h[2] = (__bf16)f0.z; h[3] = (__bf16)f0.w;
  h[4] = (__bf16)f1.x; h[5] = (__bf16)f1.y; h[6] = (__bf16)f1.z; h[7] = (__bf16)f1.w;
  return h;
}

// ---------------------------------------------------------------------------
// LayerNorm: one block (256 threads) per row of D=1024.
// ---------------------------------------------------------------------------
__global__ void __launch_bounds__(256)
ln_kernel(const float* __restrict__ x, const float* __restrict__ sc,
          const float* __restrict__ bi, float* __restrict__ out, int D) {
  __shared__ float red[256];
  const int tid = threadIdx.x;
  const long row = blockIdx.x;
  const float* xr = x + row * (long)D;

  float s = 0.0f;
  for (int i = tid; i < D; i += 256) s += xr[i];
  red[tid] = s; __syncthreads();
  for (int o = 128; o > 0; o >>= 1) { if (tid < o) red[tid] += red[tid + o]; __syncthreads(); }
  const float mu = red[0] / (float)D;
  __syncthreads();

  float v = 0.0f;
  for (int i = tid; i < D; i += 256) { float d = xr[i] - mu; v += d * d; }
  red[tid] = v; __syncthreads();
  for (int o = 128; o > 0; o >>= 1) { if (tid < o) red[tid] += red[tid + o]; __syncthreads(); }
  const float inv = rsqrtf(red[0] / (float)D + 1e-5f);

  float* orow = out + row * (long)D;
  for (int i = tid; i < D; i += 256)
    orow[i] = (xr[i] - mu) * inv * sc[i] + bi[i];
}

// ---------------------------------------------------------------------------
// GEMM:  C[m][n] = sum_k A[m][k] * W[n][k]   (+ epilogue)
// A: [M,K] f32 row-major, W: [N,K] f32 row-major.
// Tiles: BM=128, BN=64, BK=64.  256 threads = 8 waves; wave w owns a 2x2 grid
// of 16x16 tiles: row tiles 2*(w&3)+{0,1}, col tiles 2*(w>>2)+{0,1}
// -> 8 WMMAs per wave per K slab.  f32 -> bf16 at the LDS staging step.
// blockIdx.z batches experts via strides.
// ---------------------------------------------------------------------------
enum { EPI_NONE = 0, EPI_RESID = 1, EPI_BIAS = 2, EPI_BIAS_GELU = 3 };

template <int EPI>
__global__ void __launch_bounds__(256)
gemm_wmma(const float* __restrict__ A, const float* __restrict__ W,
          const float* __restrict__ bias, const float* __restrict__ R,
          float* __restrict__ C, int N, int K,
          long sA, long sW, long sB, long sC) {
  __shared__ __attribute__((aligned(16))) __bf16 As[128 * 64];  // [m][k]
  __shared__ __attribute__((aligned(16))) __bf16 Bs[64 * 64];   // [n][k]

  const int tid = threadIdx.x, lane = tid & 31, wave = tid >> 5;
  const int rp = wave & 3, cp = wave >> 2;
  const int m0 = blockIdx.y * 128, n0 = blockIdx.x * 64;
  const long z = blockIdx.z;
  A += z * sA; W += z * sW; C += z * sC;
  if (EPI == EPI_BIAS || EPI == EPI_BIAS_GELU) bias += z * sB;

  const int half8 = (lane >= 16) ? 8 : 0;     // A/C row-half select
  const int koff  = (lane >= 16) ? 8 : 0;     // B k-half select
  const int nl    = lane & 15;
  const int am0 = (rp * 2) * 16 + nl;         // A frag rows for this wave
  const int bn0 = (cp * 2) * 16 + nl;         // B frag cols for this wave

  // staging assignments
  const int arow = tid >> 1, acol = (tid & 1) * 32;   // 32 floats per thread
  const int brow = tid >> 2, bcol = (tid & 3) * 16;   // 16 floats per thread

  v8f acc[2][2] = {{{}, {}}, {{}, {}}};

  for (int kt = 0; kt < K; kt += 64) {
    __syncthreads();
    const float* ap = A + (long)(m0 + arow) * K + kt + acol;
    const float* wp = W + (long)(n0 + brow) * K + kt + bcol;
    __bf16* asd = &As[arow * 64 + acol];
    __bf16* bsd = &Bs[brow * 64 + bcol];
#pragma unroll
    for (int c = 0; c < 4; ++c) {
      float4 f0 = *(const float4*)(ap + c * 8);
      float4 f1 = *(const float4*)(ap + c * 8 + 4);
      *(v8bf*)(asd + c * 8) = cvt8(f0, f1);
    }
#pragma unroll
    for (int c = 0; c < 2; ++c) {
      float4 f0 = *(const float4*)(wp + c * 8);
      float4 f1 = *(const float4*)(wp + c * 8 + 4);
      *(v8bf*)(bsd + c * 8) = cvt8(f0, f1);
    }
    // prefetch next K slab (speculative; dropped if out of range)
    __builtin_prefetch(ap + 64, 0, 0);
    __builtin_prefetch(wp + 64, 0, 0);
    __syncthreads();

#pragma unroll
    for (int ks = 0; ks < 64; ks += 32) {
      v16bf a0 = cat8(*(const v8bf*)&As[am0 * 64 + ks + half8],
                      *(const v8bf*)&As[am0 * 64 + ks + 16 + half8]);
      v16bf a1 = cat8(*(const v8bf*)&As[(am0 + 16) * 64 + ks + half8],
                      *(const v8bf*)&As[(am0 + 16) * 64 + ks + 16 + half8]);
      v16bf b0 = cat8(*(const v8bf*)&Bs[bn0 * 64 + ks + koff],
                      *(const v8bf*)&Bs[bn0 * 64 + ks + koff + 16]);
      v16bf b1 = cat8(*(const v8bf*)&Bs[(bn0 + 16) * 64 + ks + koff],
                      *(const v8bf*)&Bs[(bn0 + 16) * 64 + ks + koff + 16]);
      acc[0][0] = WMMA_BF16(a0, b0, acc[0][0]);
      acc[0][1] = WMMA_BF16(a0, b1, acc[0][1]);
      acc[1][0] = WMMA_BF16(a1, b0, acc[1][0]);
      acc[1][1] = WMMA_BF16(a1, b1, acc[1][1]);
    }
  }

#pragma unroll
  for (int i = 0; i < 2; ++i) {
#pragma unroll
    for (int r = 0; r < 8; ++r) {
      const int gm = m0 + (rp * 2 + i) * 16 + r + half8;
#pragma unroll
      for (int j = 0; j < 2; ++j) {
        const int gn = n0 + (cp * 2 + j) * 16 + nl;
        float v = acc[i][j][r];
        if (EPI == EPI_RESID) v += R[(long)gm * N + gn];
        if (EPI == EPI_BIAS || EPI == EPI_BIAS_GELU) v += bias[gn];
        if (EPI == EPI_BIAS_GELU)
          v = 0.5f * v * (1.0f + erff(v * 0.70710678118654752f));
        C[(long)gm * N + gn] = v;
      }
    }
  }
}

// ---------------------------------------------------------------------------
// Flash attention, hd=32, H=32.  Block=128 threads (4 waves); wave owns a
// 16-row q tile.  Block stages 32x32 K/V tiles in LDS; per 32-col step each
// wave does 2 score WMMAs, online-softmax (shfl over the 16 lanes of a row),
// P -> LDS -> A-fragment relayout, then 2 PxV WMMAs.
// ---------------------------------------------------------------------------
__global__ void __launch_bounds__(128)
attn_wmma(const float* __restrict__ Q, const float* __restrict__ Km,
          const float* __restrict__ V, float* __restrict__ O, int S, int D) {
  __shared__ __attribute__((aligned(16))) __bf16 Ks[32 * 32];    // [s][k]
  __shared__ __attribute__((aligned(16))) __bf16 Vs[32 * 32];    // [n][s]
  __shared__ __attribute__((aligned(16))) __bf16 Ps[4][16 * 32]; // per-wave P

  const int tid = threadIdx.x, lane = tid & 31, wave = tid >> 5;
  const long base = ((long)blockIdx.z * S) * D + blockIdx.y * 32;
  const int half8 = (lane >= 16) ? 8 : 0;
  const int koff  = (lane >= 16) ? 8 : 0;
  const int nl    = lane & 15;
  const float scale = 0.17677669529663687f;  // 1/sqrt(32)

  // q fragment (A layout 16x32), scale folded into conversion
  const int qrow0 = blockIdx.x * 64 + wave * 16;
  const float* qp = Q + base + (long)(qrow0 + nl) * D;
  v16bf qf;
#pragma unroll
  for (int j = 0; j < 8; ++j) qf[j] = (__bf16)(qp[half8 + j] * scale);
#pragma unroll
  for (int j = 0; j < 8; ++j) qf[8 + j] = (__bf16)(qp[16 + half8 + j] * scale);

  v8f o0 = {}; v8f o1 = {};
  float mi[8], li[8];
#pragma unroll
  for (int r = 0; r < 8; ++r) { mi[r] = -1e30f; li[r] = 0.0f; }

  const int fr = tid >> 2;         // 0..31 (s row within tile)
  const int fc = (tid & 3) * 8;    // 0,8,16,24
  __bf16* Pw = Ps[wave];

  for (int s0 = 0; s0 < S; s0 += 32) {
    __syncthreads();
    const float* kp = Km + base + (long)(s0 + fr) * D + fc;
    const float* vp = V  + base + (long)(s0 + fr) * D + fc;
    __bf16* kd = &Ks[fr * 32 + fc];
#pragma unroll
    for (int j = 0; j < 8; ++j) kd[j] = (__bf16)kp[j];
#pragma unroll
    for (int j = 0; j < 8; ++j) Vs[(fc + j) * 32 + fr] = (__bf16)vp[j];
    __syncthreads();

    v16bf k0 = cat8(*(const v8bf*)&Ks[nl * 32 + koff],
                    *(const v8bf*)&Ks[nl * 32 + koff + 16]);
    v16bf k1 = cat8(*(const v8bf*)&Ks[(16 + nl) * 32 + koff],
                    *(const v8bf*)&Ks[(16 + nl) * 32 + koff + 16]);
    v8f zero = {};
    v8f sa = WMMA_BF16(qf, k0, zero);
    v8f sb = WMMA_BF16(qf, k1, zero);

    float pa[8], pb[8];
#pragma unroll
    for (int r = 0; r < 8; ++r) {
      float mx = fmaxf(sa[r], sb[r]);
#pragma unroll
      for (int msk = 1; msk < 16; msk <<= 1) mx = fmaxf(mx, __shfl_xor(mx, msk, 32));
      const float nm = fmaxf(mi[r], mx);
      const float corr = __expf(mi[r] - nm);
      pa[r] = __expf(sa[r] - nm);
      pb[r] = __expf(sb[r] - nm);
      float rs = pa[r] + pb[r];
#pragma unroll
      for (int msk = 1; msk < 16; msk <<= 1) rs += __shfl_xor(rs, msk, 32);
      li[r] = li[r] * corr + rs;
      mi[r] = nm;
      o0[r] *= corr;
      o1[r] *= corr;
    }

#pragma unroll
    for (int r = 0; r < 8; ++r) {
      const int m = r + half8;
      Pw[m * 32 + nl]      = (__bf16)pa[r];
      Pw[m * 32 + 16 + nl] = (__bf16)pb[r];
    }
    asm volatile("s_wait_dscnt 0" ::: "memory");  // in-wave LDS RAW fence

    v16bf pf = cat8(*(const v8bf*)&Pw[nl * 32 + half8],
                    *(const v8bf*)&Pw[nl * 32 + 16 + half8]);
    v16bf v0 = cat8(*(const v8bf*)&Vs[nl * 32 + koff],
                    *(const v8bf*)&Vs[nl * 32 + koff + 16]);
    v16bf v1 = cat8(*(const v8bf*)&Vs[(16 + nl) * 32 + koff],
                    *(const v8bf*)&Vs[(16 + nl) * 32 + koff + 16]);
    o0 = WMMA_BF16(pf, v0, o0);
    o1 = WMMA_BF16(pf, v1, o1);
  }

#pragma unroll
  for (int r = 0; r < 8; ++r) {
    const int gm = qrow0 + r + half8;
    const float inv = 1.0f / li[r];
    O[base + (long)gm * D + nl]      = o0[r] * inv;
    O[base + (long)gm * D + 16 + nl] = o1[r] * inv;
  }
}

// ---------------------------------------------------------------------------
extern "C" void kernel_launch(void* const* d_in, const int* in_sizes, int n_in,
                              void* d_out, int out_size, void* d_ws, size_t ws_size,
                              hipStream_t stream) {
  (void)in_sizes; (void)n_in; (void)out_size; (void)ws_size;
  const float* x    = (const float*)d_in[0];
  const float* ln1s = (const float*)d_in[1];
  const float* ln1b = (const float*)d_in[2];
  const float* wq1  = (const float*)d_in[3];
  const float* wk1  = (const float*)d_in[4];
  const float* wv1  = (const float*)d_in[5];
  const float* wo1  = (const float*)d_in[6];
  /* d_in[7] = gate_w: logits/top-k are computed in the reference but never
     consumed downstream -> no effect on the output, skipped. */
  const float* ew1  = (const float*)d_in[8];
  const float* eb1  = (const float*)d_in[9];
  const float* ew2  = (const float*)d_in[10];
  const float* eb2  = (const float*)d_in[11];
  const float* ln2s = (const float*)d_in[12];
  const float* ln2b = (const float*)d_in[13];
  const float* wq2  = (const float*)d_in[14];
  const float* wk2  = (const float*)d_in[15];
  const float* wv2  = (const float*)d_in[16];
  const float* wo2  = (const float*)d_in[17];
  float* out = (float*)d_out;

  const int B = 4, S = 1024, D = 1024, H = 32, E = 8, F = 4096;
  const int M = B * S;               // 4096 tokens
  const long MD = (long)M * D;       // 4 Mi floats

  float* ws = (float*)d_ws;
  float* xn = ws;            // 4M
  float* Qb = xn + MD;       // 4M
  float* Kb = Qb + MD;       // 4M
  float* Vb = Kb + MD;       // 4M
  float* Ob = Vb + MD;       // 4M
  float* yb = Ob + MD;       // 4M
  float* zb = yb + MD;       // 4M
  float* hb = xn;            // 16M alias over xn..Vb (dead during expert phase)

  const dim3 blk(256);
  const dim3 gP(D / 64, M / 128, 1);            // projection GEMMs
  const dim3 gAttn(S / 64, H, B);

  // ---- layer 1: PreLN MHA ----
  ln_kernel<<<M, 256, 0, stream>>>(x, ln1s, ln1b, xn, D);
  gemm_wmma<EPI_NONE><<<gP, blk, 0, stream>>>(xn, wq1, nullptr, nullptr, Qb, D, D, 0, 0, 0, 0);
  gemm_wmma<EPI_NONE><<<gP, blk, 0, stream>>>(xn, wk1, nullptr, nullptr, Kb, D, D, 0, 0, 0, 0);
  gemm_wmma<EPI_NONE><<<gP, blk, 0, stream>>>(xn, wv1, nullptr, nullptr, Vb, D, D, 0, 0, 0, 0);
  attn_wmma<<<gAttn, dim3(128), 0, stream>>>(Qb, Kb, Vb, Ob, S, D);
  gemm_wmma<EPI_RESID><<<gP, blk, 0, stream>>>(Ob, wo1, nullptr, x, yb, D, D, 0, 0, 0, 0);

  // ---- MoE expert FFN: tokens chunked [2 microbatches][8 experts][256][D] ----
  for (int m = 0; m < 2; ++m) {
    const float* Am = yb + (long)m * 2048 * D;
    float* Hm = hb + (long)m * 2048 * F;
    gemm_wmma<EPI_BIAS_GELU><<<dim3(F / 64, 256 / 128, E), blk, 0, stream>>>(
        Am, ew1, eb1, nullptr, Hm, F, D,
        (long)256 * D, (long)F * D, (long)F, (long)256 * F);
  }
  for (int m = 0; m < 2; ++m) {
    const float* Hm = hb + (long)m * 2048 * F;
    float* Zm = zb + (long)m * 2048 * D;
    gemm_wmma<EPI_BIAS><<<dim3(D / 64, 256 / 128, E), blk, 0, stream>>>(
        Hm, ew2, eb2, nullptr, Zm, D, F,
        (long)256 * F, (long)D * F, (long)D, (long)256 * D);
  }

  // ---- layer 2: PreLN MHA on z ----
  ln_kernel<<<M, 256, 0, stream>>>(zb, ln2s, ln2b, xn, D);
  gemm_wmma<EPI_NONE><<<gP, blk, 0, stream>>>(xn, wq2, nullptr, nullptr, Qb, D, D, 0, 0, 0, 0);
  gemm_wmma<EPI_NONE><<<gP, blk, 0, stream>>>(xn, wk2, nullptr, nullptr, Kb, D, D, 0, 0, 0, 0);
  gemm_wmma<EPI_NONE><<<gP, blk, 0, stream>>>(xn, wv2, nullptr, nullptr, Vb, D, D, 0, 0, 0, 0);
  attn_wmma<<<gAttn, dim3(128), 0, stream>>>(Qb, Kb, Vb, Ob, S, D);
  gemm_wmma<EPI_RESID><<<gP, blk, 0, stream>>>(Ob, wo2, nullptr, zb, out, D, D, 0, 0, 0, 0);
}
